// GraphAttentionLayer_85332410237412
// MI455X (gfx1250) — matrix operand order repaired and checked by reading, not compiled, not run on hardware
//
#include <hip/hip_runtime.h>
#include <hip/hip_bf16.h>
#include <cstdint>
#include <cstddef>

// ---------------------------------------------------------------------------
// GAT layer for MI455X (gfx1250, wave32).
//   K1: Wh = h @ W  (bf16 WMMA, fp32 acc) + s1/s2 row dots + bf16 Wh^T store
//   K2: fused masked-softmax(attn) @ Wh  (flash-style online softmax,
//       bf16 WMMA, fp32 acc), ELU epilogue.
//       B-operand chunks (shared by the 4 waves of a workgroup) are DMA'd
//       into LDS by the Tensor Data Mover (double-buffered, TENSORcnt),
//       with TDM LDS-padding for bank-conflict-free ds gathers.
// adj stream (256 MB) is the bandwidth floor (~11 us @ 23.3 TB/s); fusion
// avoids ever materializing the 16x2048x2048 score matrix.
// ---------------------------------------------------------------------------

typedef __attribute__((ext_vector_type(16))) __bf16 v16bf;
typedef __attribute__((ext_vector_type(2)))  __bf16 v2bf;
typedef __attribute__((ext_vector_type(8)))  float  v8f;

#define GAT_ALPHA 0.2f

constexpr int B_   = 16;
constexpr int N_   = 2048;
constexpr int FIN  = 256;
constexpr int FOUT = 128;
constexpr int WT_PITCH = 260;   // padded K-pitch for K1 LDS W tile

#if __has_builtin(__builtin_amdgcn_tensor_load_to_lds) && \
    __has_builtin(__builtin_amdgcn_s_wait_tensorcnt)
#define GAT_HAVE_TDM 1
#warning "GAT: TDM tensor_load_to_lds path ENABLED (async tensor DMA in K2)"
typedef __attribute__((ext_vector_type(4))) unsigned int u32x4;
typedef __attribute__((ext_vector_type(4))) int          i32x4;
typedef __attribute__((ext_vector_type(8))) int          i32x8;
#else
#define GAT_HAVE_TDM 0
#warning "GAT: TDM builtin UNAVAILABLE - cooperative LDS staging fallback in K2"
#endif

// ---------------------------------------------------------------------------
// Kernel 1: Wh = h @ W (16-row tile per wave), s1/s2, bf16 Wh^T store.
// A operand (16x32 bf16, ISA layout): lane L -> row L%16, half=L/16,
//   VGPR v<4: K = 2v + 8*half ; v>=4: K = 2v + 8 + 8*half  (pairs K,K+1)
// B operand (32x16 bf16): lane L -> col L%16, half=L/16, VGPR v: same K map as A
// C/D (16x16 f32): lane L -> col L%16, VGPR r -> row r + 8*(L/16)
// ---------------------------------------------------------------------------
__global__ __launch_bounds__(128)
void gat_wh_kernel(const float* __restrict__ h, const float* __restrict__ W,
                   const float* __restrict__ a, __bf16* __restrict__ Wht,
                   float* __restrict__ s1, float* __restrict__ s2)
{
    __shared__ __bf16 Wt[FOUT * WT_PITCH];   // W transposed [col][k], bf16, padded

    const int tid = threadIdx.x;
    for (int i = tid; i < FOUT * FIN; i += 128) {
        const int col = i >> 8;          // FIN == 256
        const int k   = i & (FIN - 1);
        Wt[col * WT_PITCH + k] = (__bf16)W[k * FOUT + col];
    }
    __syncthreads();

    const int lane = tid & 31;
    const int wave = tid >> 5;
    const int mrow = lane & 15;
    const int half = lane >> 4;
    const int tile = blockIdx.x * 4 + wave;   // 16-row tile over flattened B*N
    const int R    = tile * 16;

    float av1[8], av2[8];
#pragma unroll
    for (int t = 0; t < 8; ++t) {
        av1[t] = a[t * 16 + mrow];
        av2[t] = a[FOUT + t * 16 + mrow];
    }

    v8f acc[8] = {};
    const int arow = R + mrow;

#pragma unroll 1
    for (int kc = 0; kc < FIN / 32; ++kc) {
        v16bf A;
#pragma unroll
        for (int v = 0; v < 8; ++v) {
            const int k = kc * 32 + ((v < 4) ? (2 * v + 8 * half)
                                             : (2 * v + 8 + 8 * half));
            const float2 f2 = *(const float2*)(h + (size_t)arow * FIN + k);
            A[2 * v]     = (__bf16)f2.x;
            A[2 * v + 1] = (__bf16)f2.y;
        }
#pragma unroll
        for (int t = 0; t < 8; ++t) {
            v16bf Bv;
#pragma unroll
            for (int v = 0; v < 8; ++v) {
                const int k = kc * 32 + ((v < 4) ? (2 * v + 8 * half) : (2 * v + 8 + 8 * half));
                const v2bf pr = *(const v2bf*)&Wt[(t * 16 + mrow) * WT_PITCH + k];
                Bv[2 * v]     = pr.x;
                Bv[2 * v + 1] = pr.y;
            }
            acc[t] = __builtin_amdgcn_wmma_f32_16x16x32_bf16(
                         false, A, false, Bv, (short)0, acc[t], false, false);
        }
    }

    // epilogue: s1/s2 row dots (butterfly over the 16-lane half sharing the
    // C-layout row), plus bf16 Wh^T store [b][col][n]
    const int b    = R / N_;
    const int nbas = R - b * N_;
#pragma unroll
    for (int r = 0; r < 8; ++r) {
        const int grow = R + r + 8 * half;
        float p1 = 0.f, p2 = 0.f;
#pragma unroll
        for (int t = 0; t < 8; ++t) {
            const float v = acc[t][r];
            p1 += v * av1[t];
            p2 += v * av2[t];
            Wht[(size_t)(b * FOUT + t * 16 + mrow) * N_ + (nbas + r + 8 * half)] =
                (__bf16)v;
        }
#pragma unroll
        for (int off = 1; off < 16; off <<= 1) {
            p1 += __shfl_xor(p1, off);
            p2 += __shfl_xor(p2, off);
        }
        if (mrow == 0) { s1[grow] = p1; s2[grow] = p2; }
    }
}

// ---------------------------------------------------------------------------
// TDM: DMA one 32(n) x 128(col) bf16 tile of Wh^T into LDS.
// D# per CDNA5 ISA:
//   g0: count=1 | lds_addr[63:32] | global_addr[120:64] | type=2[127:126]
//   g1: data_size=2B, pad_enable, pad_interval=16 DW (one 64B row),
//       pad_amount=1 DW  -> LDS pitch 68 B/col, 17 mod 64 -> conflict-free;
//       tensor_dim0=2048, tensor_dim1=128, tile_dim0=32, tile_dim1=128,
//       tensor_dim0_stride=2048
// ---------------------------------------------------------------------------
constexpr int BT_PITCH_B = 68;                 // 64B row + 4B TDM pad
constexpr int BT_BUF_B   = FOUT * BT_PITCH_B;  // 8704 B per chunk buffer

#if GAT_HAVE_TDM
__device__ __forceinline__ void tdm_stage_bt(const __bf16* gsrc,
                                             unsigned int lds_byte)
{
    const unsigned long long ga = (unsigned long long)(uintptr_t)gsrc;
    u32x4 g0;
    g0[0] = 1u;                                            // count=1, user D#
    g0[1] = lds_byte;                                      // LDS dest offset
    g0[2] = (unsigned int)ga;                              // global_addr lo
    g0[3] = (unsigned int)((ga >> 32) & 0x1FFFFFFull)      // global_addr hi
          | (2u << 30);                                    // type = 2 (image)
    i32x8 g1;
    g1[0] = (1 << 16) | (1 << 20) | (3 << 22);  // dsz=2B | pad_en | interval=16DW
    g1[1] = (int)(2048u << 16);                 // tensor_dim0 = 2048 (lo 16)
    g1[2] = (int)(128u << 16);                  // td0 hi=0 | tensor_dim1 = 128
    g1[3] = (int)(32u << 16);                   // td1 hi=0 | tile_dim0 = 32
    g1[4] = 128;                                // tile_dim1 = 128, tile_dim2 = 0
    g1[5] = 2048;                               // tensor_dim0_stride lo
    g1[6] = 0;                                  // stride0 hi | stride1 lo
    g1[7] = 0;                                  // stride1 hi (unused, 2D)
    const i32x4 z4 = {0, 0, 0, 0};
#if defined(__clang_major__) && (__clang_major__ >= 23)
    const i32x8 z8 = {0, 0, 0, 0, 0, 0, 0, 0};
    __builtin_amdgcn_tensor_load_to_lds(g0, g1, z4, z4, z8, 0);
#else
    __builtin_amdgcn_tensor_load_to_lds(g0, g1, z4, z4, 0);
#endif
}
#endif

// ---------------------------------------------------------------------------
// Kernel 2: fused masked-softmax + attn @ Wh, 16-row tile per wave,
// online (flash) softmax over j in chunks of 32 (= WMMA K).
// The 4 waves of a workgroup share batch b; the per-chunk Wh^T B-tile is
// staged once into LDS (TDM double-buffered when available).
// ---------------------------------------------------------------------------
__global__ __launch_bounds__(128)
void gat_attn_kernel(const int* __restrict__ adj, const __bf16* __restrict__ Wht,
                     const float* __restrict__ s1, const float* __restrict__ s2,
                     float* __restrict__ out)
{
    __shared__ unsigned int BtU[2 * BT_BUF_B / 4];   // 2 x 8704 B chunk buffers

    const int tid  = threadIdx.x;
    const int lane = tid & 31;
    const int wave = tid >> 5;
    const int mrow = lane & 15;
    const int half = lane >> 4;
    const int tile = blockIdx.x * 4 + wave;
    const int b    = tile / (N_ / 16);       // 4 tiles/block all in same batch
    const int rloc = (tile % (N_ / 16)) * 16;

    const float s1v   = s1[(size_t)b * N_ + rloc + mrow];
    const float* s2b  = s2 + (size_t)b * N_;
    const int* adjrow = adj + ((size_t)b * N_ + rloc + mrow) * N_;
    const __bf16* wbase = Wht + (size_t)b * FOUT * N_;   // Wh^T [col][n]

    const unsigned int lds_base = (unsigned int)(size_t)(void*)BtU;
    (void)lds_base;

#if GAT_HAVE_TDM
    if (wave == 0) tdm_stage_bt(wbase, lds_base);        // chunk 0 -> buf 0
#endif

    float m = -INFINITY, l = 0.f;
    v8f acc[8] = {};

#pragma unroll 1
    for (int j0 = 0, c = 0; j0 < N_; j0 += 32, ++c) {
        const int cur = c & 1;
#if GAT_HAVE_TDM
        if (wave == 0) __builtin_amdgcn_s_wait_tensorcnt(0);  // chunk c landed
        __syncthreads();                     // all waves: buf[cur] ready,
                                             // buf[cur^1] readers done
        if ((j0 + 32 < N_) && (wave == 0))   // overlap DMA of c+1 with compute
            tdm_stage_bt(wbase + (j0 + 32),
                         lds_base + (unsigned)(cur ^ 1) * BT_BUF_B);
        const char* bb = (const char*)BtU + cur * BT_BUF_B;
#else
        __syncthreads();                     // buf readers of chunk c-1 done
        for (int pi = tid; pi < 2048; pi += 128) {        // coop stage chunk c
            const int col = pi >> 4, dw = pi & 15;
            BtU[col * 17 + dw] =
                *(const unsigned int*)(wbase + (size_t)col * N_ + j0 + dw * 2);
        }
        __syncthreads();
        const char* bb = (const char*)BtU;
        (void)cur;
#endif
        const float s2reg = s2b[j0 + lane];          // 32 scores across wave

        // adjacency for this lane's 16 (row,k) slots; matches A-operand K map:
        // half=0 -> k in {0..7, 16..23}, half=1 -> k in {8..15, 24..31}
        const int4 q0 = *(const int4*)(adjrow + j0 + 8 * half);
        const int4 q1 = *(const int4*)(adjrow + j0 + 8 * half + 4);
        const int4 q2 = *(const int4*)(adjrow + j0 + 16 + 8 * half);
        const int4 q3 = *(const int4*)(adjrow + j0 + 16 + 8 * half + 4);
        if (j0 + 32 < N_) {                           // stream next adj chunk
            __builtin_prefetch(adjrow + j0 + 32 + 8 * half, 0, 0);
            __builtin_prefetch(adjrow + j0 + 48 + 8 * half, 0, 0);
        }
        const int am[16] = { q0.x, q0.y, q0.z, q0.w, q1.x, q1.y, q1.z, q1.w,
                             q2.x, q2.y, q2.z, q2.w, q3.x, q3.y, q3.z, q3.w };

        float e[16];
        float emax = -INFINITY;
#pragma unroll
        for (int i = 0; i < 16; ++i) {
            const int k = (i < 8) ? (8 * half + i) : (16 + 8 * half + (i - 8));
            float ev = s1v + __shfl(s2reg, k);
            ev = (ev >= 0.f) ? ev : GAT_ALPHA * ev;   // leaky relu, then mask
            e[i] = ev;
            emax = (am[i] > 0) ? fmaxf(emax, ev) : emax;
        }

        const float cm    = fmaxf(emax, __shfl_xor(emax, 16)); // combine halves
        const float newm  = fmaxf(m, cm);
        const float scale = (newm == -INFINITY) ? 1.f : __expf(m - newm);

        float lpart = 0.f;
        v16bf A;                 // A element index == local-k enumeration above
#pragma unroll
        for (int i = 0; i < 16; ++i) {
            const float p = (am[i] > 0) ? __expf(e[i] - newm) : 0.f;
            lpart += p;
            A[i] = (__bf16)p;
        }
        l = l * scale + lpart + __shfl_xor(lpart, 16);
        m = newm;

        // per-C-row rescale: row r+8*half's scale lives in lane r+8*half
        v8f rs;
#pragma unroll
        for (int r = 0; r < 8; ++r) rs[r] = __shfl(scale, r + 8 * half);

#pragma unroll
        for (int t = 0; t < 8; ++t) {
            v16bf Bv;
            const char* wp = bb + (t * 16 + mrow) * BT_PITCH_B;
#pragma unroll
            for (int v = 0; v < 8; ++v) {
                const v2bf pr = *(const v2bf*)(wp + ((v < 4) ? (2 * v + 8 * half) : (2 * v + 8 + 8 * half)) * 2);
                Bv[2 * v]     = pr.x;
                Bv[2 * v + 1] = pr.y;
            }
            acc[t] = acc[t] * rs;
            acc[t] = __builtin_amdgcn_wmma_f32_16x16x32_bf16(
                         false, A, false, Bv, (short)0, acc[t], false, false);
        }
    }

    // epilogue: normalize, ELU, store
#pragma unroll
    for (int r = 0; r < 8; ++r) {
        const float lr  = __shfl(l, r + 8 * half);
        const float inv = 1.f / lr;
        const size_t ob = ((size_t)b * N_ + rloc + r + 8 * half) * FOUT;
#pragma unroll
        for (int t = 0; t < 8; ++t) {
            float v = acc[t][r] * inv;
            v = (v > 0.f) ? v : (__expf(v) - 1.f);
            out[ob + t * 16 + mrow] = v;
        }
    }
}

// ---------------------------------------------------------------------------
extern "C" void kernel_launch(void* const* d_in, const int* in_sizes, int n_in,
                              void* d_out, int out_size, void* d_ws, size_t ws_size,
                              hipStream_t stream)
{
    (void)in_sizes; (void)n_in; (void)out_size; (void)ws_size;

    const float* h   = (const float*)d_in[0];   // (16, 2048, 256) f32
    const int*   adj = (const int*)  d_in[1];   // (16, 2048, 2048) i32
    const float* W   = (const float*)d_in[2];   // (256, 128) f32
    const float* a   = (const float*)d_in[3];   // (256, 1) f32
    float* out = (float*)d_out;                 // (16, 2048, 128) f32

    // workspace: Wh^T bf16 [B][FOUT][N] (8 MB) + s1,s2 f32 [B*N] (128 KB each)
    __bf16* Wht = (__bf16*)d_ws;
    float*  s1  = (float*)((char*)d_ws + (size_t)B_ * FOUT * N_ * sizeof(__bf16));
    float*  s2  = s1 + (size_t)B_ * N_;

    const dim3 blk(128);                         // 4 waves, 1 tile each
    const dim3 grd((B_ * N_ / 16) / 4);          // 512 blocks

    hipLaunchKernelGGL(gat_wh_kernel,   grd, blk, 0, stream, h, W, a, Wht, s1, s2);
    hipLaunchKernelGGL(gat_attn_kernel, grd, blk, 0, stream, adj, Wht, s1, s2, out);
}
